// ModularMultiHeadAttention_22333829939869
// MI455X (gfx1250) — matrix operand order repaired and checked by reading, not compiled, not run on hardware
//
#include <hip/hip_runtime.h>
#include <hip/hip_bf16.h>

// ---------------------------------------------------------------------------
// ModularMultiHeadAttention for MI455X (gfx1250, wave32, WMMA + TDM)
//   B=1, T=2048, D=2048, H=32, HD=64
// Pipeline: cvt(fp32->bf16) -> fused QKV GEMM + RoPE (TDM-staged, LDS-padded
//           B tiles, 32x64 per wave) -> flash attention (online softmax,
//           causal+bias+module weights) -> output GEMM.
// Matrix math via v_wmma_f32_16x16x32_bf16; weight tiles via tensor_load_to_lds
// with D# pad fields for bank-conflict-free ds_load_b128.
// ---------------------------------------------------------------------------

#define TSEQ   2048
#define DMODEL 2048
#define NHEADS 32
#define HDIM   64
#define NEGINF (-1e9f)

// LDS B tile: 64 rows x 32 bf16, padded to 40-element (80B) row stride.
// 20-DWORD stride => 16 lanes' b128 groups cover all 64 banks exactly once.
#define BROW   40
#define BTILEB (HDIM * BROW * 2)   // 5120 bytes per buffer

typedef __attribute__((ext_vector_type(16))) __bf16 v16bf;
typedef __attribute__((ext_vector_type(8)))  float  v8f;
typedef __attribute__((ext_vector_type(4)))  unsigned int u32x4;
typedef __attribute__((ext_vector_type(8)))  int i32x8;
typedef __attribute__((ext_vector_type(4)))  int i32x4;

union Frag { v16bf v; uint4 q[2]; };

#if defined(__has_builtin)
#if __has_builtin(__builtin_amdgcn_tensor_load_to_lds) && \
    __has_builtin(__builtin_amdgcn_s_wait_tensorcnt)
#define HAVE_TDM 1
#endif
#endif
#ifndef HAVE_TDM
#define HAVE_TDM 0
#endif

// HEAD_COUNTS = [8,6,4,4,6,4] -> cumulative [8,14,18,22,28,32]
__device__ __forceinline__ int head2mod(int hh) {
    return (hh < 8) ? 0 : (hh < 14) ? 1 : (hh < 18) ? 2
         : (hh < 22) ? 3 : (hh < 28) ? 4 : 5;
}

#if HAVE_TDM
// TDM: async-load a 64-row x 32-col bf16 tile (row stride DMODEL) into LDS,
// inserting 4 DWORDs of padding after each 16-DWORD (64B) row => 80B stride.
// D# packing per CDNA5 ISA.
__device__ __forceinline__ void tdm_load_tile_64x32(const __bf16* gsrc,
                                                    unsigned lds_off) {
    const unsigned long long ga = (unsigned long long)(uintptr_t)gsrc;
    u32x4 g0 = { 1u, lds_off, (unsigned)ga,
                 (unsigned)((ga >> 32) & 0x01FFFFFFull) | 0x80000000u };
    i32x8 g1 = { 0x06D10000,          // data_size=2B | pad_en | intv=16DW | amt=4DW
                 0x08000000,          // tensor_dim0[15:0]=2048 << 16
                 0x08000000,          // tensor_dim1[15:0]=2048 << 16
                 0x00200000,          // tile_dim0=32 << 16
                 0x00000040,          // tile_dim1=64
                 0x00000800,          // tensor_dim0_stride=2048
                 0, 0 };
    i32x4 g2 = { 0, 0, 0, 0 };
    i32x4 g3 = { 0, 0, 0, 0 };
#if __clang_major__ >= 23
    i32x8 gz = { 0, 0, 0, 0, 0, 0, 0, 0 };
    __builtin_amdgcn_tensor_load_to_lds(g0, g1, g2, g3, gz, 0);
#else
    __builtin_amdgcn_tensor_load_to_lds(g0, g1, g2, g3, 0);
#endif
}
#endif

// ------------------------- fp32 -> bf16 conversion -------------------------
__global__ __launch_bounds__(256) void k_cvt_bf16(const float* __restrict__ s,
                                                  __bf16* __restrict__ d, int n) {
    int i = (blockIdx.x * blockDim.x + threadIdx.x) * 8;
    const int stride = gridDim.x * blockDim.x * 8;
    for (; i < n; i += stride) {
        float4 f0 = *(const float4*)(s + i);
        float4 f1 = *(const float4*)(s + i + 4);
        alignas(16) __bf16 t[8] = {
            (__bf16)f0.x, (__bf16)f0.y, (__bf16)f0.z, (__bf16)f0.w,
            (__bf16)f1.x, (__bf16)f1.y, (__bf16)f1.z, (__bf16)f1.w };
        *(uint4*)(d + i) = *(const uint4*)t;
    }
}

// --------------------- fused QKV projection + RoPE -------------------------
// grid: (T/32/8, NHEADS, 3[q,k,v]); block: 256 (8 waves);
// wave = one 32x64 output tile (2 M-subtiles); B tile TDM double-buffered.
__global__ __launch_bounds__(256) void k_qkv_rope(
    const __bf16* __restrict__ xb,
    const __bf16* __restrict__ Wqb, const __bf16* __restrict__ Wkb,
    const __bf16* __restrict__ Wvb,
    __bf16* __restrict__ Qh, __bf16* __restrict__ Kh, __bf16* __restrict__ Vt)
{
    __shared__ alignas(16) __bf16 ldsB[2][HDIM * BROW];   // 2 x 5 KB

    const int wave = threadIdx.x >> 5;
    const int lane = threadIdx.x & 31;
    const int h  = lane >> 4;
    const int ln = lane & 15;
    const int m0   = (blockIdx.x * 8 + wave) * 32;        // 32 sequence rows
    const int head = blockIdx.y;
    const int mat  = blockIdx.z;                          // 0=Q 1=K 2=V
    const __bf16* W = (mat == 0) ? Wqb : (mat == 1) ? Wkb : Wvb;
    const __bf16* Wtile = W + (size_t)head * HDIM * DMODEL;

#if HAVE_TDM
    const unsigned ldsBase = (unsigned)(uintptr_t)&ldsB[0][0];
    if (wave == 0) tdm_load_tile_64x32(Wtile, ldsBase);
#else
    {   // cooperative padded copy: 256 threads x 16B = 4 KB tile
        const int row = threadIdx.x >> 2, seg = threadIdx.x & 3;
        *(uint4*)&ldsB[0][row * BROW + seg * 8] =
            *(const uint4*)(Wtile + row * DMODEL + seg * 8);
    }
#endif

    Frag a[2];
#pragma unroll
    for (int mi = 0; mi < 2; ++mi) {
        const __bf16* xr = xb + (size_t)(m0 + 16 * mi + ln) * DMODEL;
        a[mi].q[0] = *(const uint4*)(xr + 16 * h);
        a[mi].q[1] = *(const uint4*)(xr + 16 * h + 8);
    }

    v8f acc[2][4] = {};
    int buf = 0;
    for (int k0 = 0; k0 < DMODEL; k0 += 32, buf ^= 1) {
#if HAVE_TDM
        if (wave == 0) __builtin_amdgcn_s_wait_tensorcnt(0);
        __syncthreads();                                  // tile[buf] published
        if (wave == 0 && k0 + 32 < DMODEL)
            tdm_load_tile_64x32(Wtile + k0 + 32, ldsBase + (buf ^ 1) * BTILEB);
#else
        __syncthreads();
        if (k0 + 32 < DMODEL) {
            const int row = threadIdx.x >> 2, seg = threadIdx.x & 3;
            *(uint4*)&ldsB[buf ^ 1][row * BROW + seg * 8] =
                *(const uint4*)(Wtile + row * DMODEL + k0 + 32 + seg * 8);
        }
#endif
        // prefetch next A fragments while current WMMAs run
        Frag an[2] = { a[0], a[1] };
        if (k0 + 32 < DMODEL) {
#pragma unroll
            for (int mi = 0; mi < 2; ++mi) {
                const __bf16* xr = xb + (size_t)(m0 + 16 * mi + ln) * DMODEL + k0 + 32;
                an[mi].q[0] = *(const uint4*)(xr + 16 * h);
                an[mi].q[1] = *(const uint4*)(xr + 16 * h + 8);
            }
        }
        // preload all 4 B fragments, then issue 8 back-to-back WMMAs
        const __bf16* Bt = &ldsB[buf][0];
        Frag b[4];
#pragma unroll
        for (int j = 0; j < 4; ++j) {
            const __bf16* br = Bt + (16 * j + ln) * BROW + 16 * h;
            b[j].q[0] = *(const uint4*)(br);
            b[j].q[1] = *(const uint4*)(br + 8);
        }
#pragma unroll
        for (int mi = 0; mi < 2; ++mi)
#pragma unroll
            for (int j = 0; j < 4; ++j)
                acc[mi][j] = __builtin_amdgcn_wmma_f32_16x16x32_bf16(
                    false, a[mi].v, false, b[j].v, (short)0, acc[mi][j], false, false);
        a[0] = an[0]; a[1] = an[1];
    }

    if (mat < 2) {
        // RoPE: col i pairs with i+-32 -> acc[.][j] with acc[.][j^2], same lane.
        const float invf0 = __expf(-0.28782313662f * (float)ln);
        const float invf1 = __expf(-0.28782313662f * (float)(16 + ln));
        __bf16* dst = ((mat == 0) ? Qh : Kh) + (size_t)head * TSEQ * HDIM;
#pragma unroll
        for (int mi = 0; mi < 2; ++mi) {
#pragma unroll
            for (int r = 0; r < 8; ++r) {
                const int t = m0 + 16 * mi + r + 8 * h;
                const float ft = (float)t;
                const float a0 = ft * invf0, a1 = ft * invf1;
                const float c0 = __cosf(a0), s0 = __sinf(a0);
                const float c1 = __cosf(a1), s1 = __sinf(a1);
                const float o0 = acc[mi][0][r] * c0 - acc[mi][2][r] * s0;
                const float o1 = acc[mi][1][r] * c1 - acc[mi][3][r] * s1;
                const float o2 = acc[mi][2][r] * c0 + acc[mi][0][r] * s0;
                const float o3 = acc[mi][3][r] * c1 + acc[mi][1][r] * s1;
                __bf16* p = dst + t * HDIM + ln;
                p[0]  = (__bf16)o0; p[16] = (__bf16)o1;
                p[32] = (__bf16)o2; p[48] = (__bf16)o3;
            }
        }
    } else {
        // V stored transposed [H][HD][T] so PV B-fragments load contiguously
#pragma unroll
        for (int mi = 0; mi < 2; ++mi)
#pragma unroll
            for (int r = 0; r < 8; ++r) {
                const int t = m0 + 16 * mi + r + 8 * h;
#pragma unroll
                for (int j = 0; j < 4; ++j)
                    Vt[((size_t)head * HDIM + 16 * j + ln) * TSEQ + t] =
                        (__bf16)acc[mi][j][r];
            }
    }
}

// --------------------------- flash attention -------------------------------
// grid: (T/16, NHEADS); block: 32 (one wave owns a 16-query tile of one head)
__global__ __launch_bounds__(32) void k_flash(
    const __bf16* __restrict__ Qh, const __bf16* __restrict__ Kh,
    const __bf16* __restrict__ Vt, const float* __restrict__ bias,
    const float* __restrict__ modw, const float* __restrict__ tscale,
    __bf16* __restrict__ Ob)
{
    __shared__ alignas(16) __bf16 ldsP[16 * 32];   // P tile, C-layout -> A-layout
    const int lane = threadIdx.x & 31;
    const int h  = lane >> 4;
    const int ln = lane & 15;
    const int q0   = blockIdx.x * 16;
    const int head = blockIdx.y;
    const float scale = 0.125f * tscale[head];     // HD^-0.5 * temp_scales
    const float w = modw[head2mod(head)];

    Frag aQ[2];                                    // Q rows, reused all blocks
    {
        const __bf16* qr = Qh + ((size_t)head * TSEQ + q0 + ln) * HDIM;
#pragma unroll
        for (int kk = 0; kk < 2; ++kk) {
            aQ[kk].q[0] = *(const uint4*)(qr + 32 * kk + 16 * h);
            aQ[kk].q[1] = *(const uint4*)(qr + 32 * kk + 16 * h + 8);
        }
    }

    float Mrow[8], Lrow[8];
#pragma unroll
    for (int r = 0; r < 8; ++r) { Mrow[r] = -3.0e38f; Lrow[r] = 0.0f; }
    v8f accO[4] = {};

    const int nkb = (q0 + 15) / 32 + 1;            // causal: keys <= q0+15
    for (int kb = 0; kb < nkb; ++kb) {
        const int key0 = kb * 32;
        // prefetch next block's bias tile (16 rows x 128B)
        if (kb + 1 < nkb)
            __builtin_prefetch(&bias[(size_t)(q0 + ln) * TSEQ + key0 + 32 + 16 * h], 0, 3);
        // ---- S = Q K^T : preload 4 B frags, then 4 WMMAs ----
        Frag bK[4];                                // [nb][kk]
#pragma unroll
        for (int nb = 0; nb < 2; ++nb) {
            const __bf16* kr = Kh + ((size_t)head * TSEQ + key0 + 16 * nb + ln) * HDIM + 16 * h;
#pragma unroll
            for (int kk = 0; kk < 2; ++kk) {
                bK[nb * 2 + kk].q[0] = *(const uint4*)(kr + 32 * kk);
                bK[nb * 2 + kk].q[1] = *(const uint4*)(kr + 32 * kk + 8);
            }
        }
        v8f s[2] = {};
#pragma unroll
        for (int nb = 0; nb < 2; ++nb)
#pragma unroll
            for (int kk = 0; kk < 2; ++kk)
                s[nb] = __builtin_amdgcn_wmma_f32_16x16x32_bf16(
                    false, aQ[kk].v, false, bK[nb * 2 + kk].v, (short)0, s[nb], false, false);
        // ---- mask + bias + module weight + online softmax ----
#pragma unroll
        for (int r = 0; r < 8; ++r) {
            const int qi  = q0 + r + 8 * h;        // C-layout row
            const int ki0 = key0 + ln, ki1 = key0 + 16 + ln;
            float v0 = s[0][r] * scale, v1 = s[1][r] * scale;
            v0 = (ki0 > qi) ? NEGINF : v0;
            v1 = (ki1 > qi) ? NEGINF : v1;
            v0 = (v0 + bias[(size_t)qi * TSEQ + ki0]) * w;
            v1 = (v1 + bias[(size_t)qi * TSEQ + ki1]) * w;
            float mx = fmaxf(v0, v1);              // row max across 16 lanes
            mx = fmaxf(mx, __shfl_xor(mx, 1, 32));
            mx = fmaxf(mx, __shfl_xor(mx, 2, 32));
            mx = fmaxf(mx, __shfl_xor(mx, 4, 32));
            mx = fmaxf(mx, __shfl_xor(mx, 8, 32));
            const float Mn = fmaxf(Mrow[r], mx);
            const float cf = __expf(Mrow[r] - Mn);
            const float p0 = __expf(v0 - Mn), p1 = __expf(v1 - Mn);
            float rs = p0 + p1;
            rs += __shfl_xor(rs, 1, 32);
            rs += __shfl_xor(rs, 2, 32);
            rs += __shfl_xor(rs, 4, 32);
            rs += __shfl_xor(rs, 8, 32);
            Lrow[r] = Lrow[r] * cf + rs;
            Mrow[r] = Mn;
#pragma unroll
            for (int j = 0; j < 4; ++j) accO[j][r] *= cf;
            ldsP[(r + 8 * h) * 32 + ln]      = (__bf16)p0;
            ldsP[(r + 8 * h) * 32 + 16 + ln] = (__bf16)p1;
        }
        __syncthreads();
        // ---- O += P V : preload A (LDS) + 4 B frags, then 4 WMMAs ----
        Frag aP;
        aP.q[0] = *(const uint4*)(&ldsP[ln * 32 + 16 * h]);
        aP.q[1] = *(const uint4*)(&ldsP[ln * 32 + 16 * h + 8]);
        Frag bV[4];
#pragma unroll
        for (int j = 0; j < 4; ++j) {
            const __bf16* vr = Vt + ((size_t)head * HDIM + 16 * j + ln) * TSEQ + key0 + 16 * h;
            bV[j].q[0] = *(const uint4*)(vr);
            bV[j].q[1] = *(const uint4*)(vr + 8);
        }
#pragma unroll
        for (int j = 0; j < 4; ++j)
            accO[j] = __builtin_amdgcn_wmma_f32_16x16x32_bf16(
                false, aP.v, false, bV[j].v, (short)0, accO[j], false, false);
        __syncthreads();
    }

    // ---- normalize and emit bf16 [T][D] for the output projection ----
#pragma unroll
    for (int r = 0; r < 8; ++r) {
        const float inv = 1.0f / Lrow[r];
        const int t = q0 + r + 8 * h;
        __bf16* p = Ob + (size_t)t * DMODEL + head * HDIM + ln;
        p[0]  = (__bf16)(accO[0][r] * inv);
        p[16] = (__bf16)(accO[1][r] * inv);
        p[32] = (__bf16)(accO[2][r] * inv);
        p[48] = (__bf16)(accO[3][r] * inv);
    }
}

// -------------------------- output projection ------------------------------
// grid: (T/32/8, D/64); block: 256 (8 waves); 32x64 per wave; TDM-staged Wo.
__global__ __launch_bounds__(256) void k_oproj(
    const __bf16* __restrict__ Ob, const __bf16* __restrict__ Wob,
    float* __restrict__ out)
{
    __shared__ alignas(16) __bf16 ldsB[2][HDIM * BROW];

    const int wave = threadIdx.x >> 5;
    const int lane = threadIdx.x & 31;
    const int h  = lane >> 4;
    const int ln = lane & 15;
    const int m0 = (blockIdx.x * 8 + wave) * 32;
    const int n0 = blockIdx.y * 64;
    const __bf16* Wtile = Wob + (size_t)n0 * DMODEL;

#if HAVE_TDM
    const unsigned ldsBase = (unsigned)(uintptr_t)&ldsB[0][0];
    if (wave == 0) tdm_load_tile_64x32(Wtile, ldsBase);
#else
    {
        const int row = threadIdx.x >> 2, seg = threadIdx.x & 3;
        *(uint4*)&ldsB[0][row * BROW + seg * 8] =
            *(const uint4*)(Wtile + row * DMODEL + seg * 8);
    }
#endif

    Frag a[2];
#pragma unroll
    for (int mi = 0; mi < 2; ++mi) {
        const __bf16* ar = Ob + (size_t)(m0 + 16 * mi + ln) * DMODEL;
        a[mi].q[0] = *(const uint4*)(ar + 16 * h);
        a[mi].q[1] = *(const uint4*)(ar + 16 * h + 8);
    }

    v8f acc[2][4] = {};
    int buf = 0;
    for (int k0 = 0; k0 < DMODEL; k0 += 32, buf ^= 1) {
#if HAVE_TDM
        if (wave == 0) __builtin_amdgcn_s_wait_tensorcnt(0);
        __syncthreads();
        if (wave == 0 && k0 + 32 < DMODEL)
            tdm_load_tile_64x32(Wtile + k0 + 32, ldsBase + (buf ^ 1) * BTILEB);
#else
        __syncthreads();
        if (k0 + 32 < DMODEL) {
            const int row = threadIdx.x >> 2, seg = threadIdx.x & 3;
            *(uint4*)&ldsB[buf ^ 1][row * BROW + seg * 8] =
                *(const uint4*)(Wtile + row * DMODEL + k0 + 32 + seg * 8);
        }
#endif
        Frag an[2] = { a[0], a[1] };
        if (k0 + 32 < DMODEL) {
#pragma unroll
            for (int mi = 0; mi < 2; ++mi) {
                const __bf16* ar = Ob + (size_t)(m0 + 16 * mi + ln) * DMODEL + k0 + 32;
                an[mi].q[0] = *(const uint4*)(ar + 16 * h);
                an[mi].q[1] = *(const uint4*)(ar + 16 * h + 8);
            }
        }
        const __bf16* Bt = &ldsB[buf][0];
        Frag b[4];
#pragma unroll
        for (int j = 0; j < 4; ++j) {
            const __bf16* br = Bt + (16 * j + ln) * BROW + 16 * h;
            b[j].q[0] = *(const uint4*)(br);
            b[j].q[1] = *(const uint4*)(br + 8);
        }
#pragma unroll
        for (int mi = 0; mi < 2; ++mi)
#pragma unroll
            for (int j = 0; j < 4; ++j)
                acc[mi][j] = __builtin_amdgcn_wmma_f32_16x16x32_bf16(
                    false, a[mi].v, false, b[j].v, (short)0, acc[mi][j], false, false);
        a[0] = an[0]; a[1] = an[1];
    }
#pragma unroll
    for (int mi = 0; mi < 2; ++mi)
#pragma unroll
        for (int r = 0; r < 8; ++r) {
            const int t = m0 + 16 * mi + r + 8 * h;
            float* p = out + (size_t)t * DMODEL + n0 + ln;
            p[0]  = acc[mi][0][r];
            p[16] = acc[mi][1][r];
            p[32] = acc[mi][2][r];
            p[48] = acc[mi][3][r];
        }
}

// ---------------------------------------------------------------------------
extern "C" void kernel_launch(void* const* d_in, const int* in_sizes, int n_in,
                              void* d_out, int out_size, void* d_ws, size_t ws_size,
                              hipStream_t stream) {
    (void)in_sizes; (void)n_in; (void)out_size; (void)ws_size;
    const float* x    = (const float*)d_in[0];
    // d_in[1]: causal mask (int) — statically known, unused
    const float* bias = (const float*)d_in[2];
    const float* modw = (const float*)d_in[3];
    const float* Wq   = (const float*)d_in[4];
    const float* Wk   = (const float*)d_in[5];
    const float* Wv   = (const float*)d_in[6];
    const float* Wo   = (const float*)d_in[7];
    const float* ts   = (const float*)d_in[8];

    char* ws = (char*)d_ws;
    const size_t MB = 1024 * 1024;
    __bf16* xb  = (__bf16*)(ws +  0 * MB);   // [T][D]     8 MB
    __bf16* Wqb = (__bf16*)(ws +  8 * MB);   // [D][D]     8 MB
    __bf16* Wkb = (__bf16*)(ws + 16 * MB);
    __bf16* Wvb = (__bf16*)(ws + 24 * MB);
    __bf16* Wob = (__bf16*)(ws + 32 * MB);
    __bf16* Qh  = (__bf16*)(ws + 40 * MB);   // [H][T][64] 8 MB
    __bf16* Kh  = (__bf16*)(ws + 48 * MB);   // [H][T][64] 8 MB
    __bf16* Vt  = (__bf16*)(ws + 56 * MB);   // [H][64][T] 8 MB (transposed)
    __bf16* Ob  = (__bf16*)(ws + 64 * MB);   // [T][D]     8 MB

    const int N = TSEQ * DMODEL;
    k_cvt_bf16<<<dim3(512), dim3(256), 0, stream>>>(x,  xb,  N);
    k_cvt_bf16<<<dim3(512), dim3(256), 0, stream>>>(Wq, Wqb, N);
    k_cvt_bf16<<<dim3(512), dim3(256), 0, stream>>>(Wk, Wkb, N);
    k_cvt_bf16<<<dim3(512), dim3(256), 0, stream>>>(Wv, Wvb, N);
    k_cvt_bf16<<<dim3(512), dim3(256), 0, stream>>>(Wo, Wob, N);

    k_qkv_rope<<<dim3(TSEQ / 32 / 8, NHEADS, 3), dim3(256), 0, stream>>>(
        xb, Wqb, Wkb, Wvb, Qh, Kh, Vt);

    k_flash<<<dim3(TSEQ / 16, NHEADS), dim3(32), 0, stream>>>(
        Qh, Kh, Vt, bias, modw, ts, Ob);

    k_oproj<<<dim3(TSEQ / 32 / 8, DMODEL / 64), dim3(256), 0, stream>>>(
        Ob, Wob, (float*)d_out);
}
